// TriangleAttentionStartingNode_41961830482537
// MI455X (gfx1250) — matrix operand order repaired and checked by reading, not compiled, not run on hardware
//
#include <hip/hip_runtime.h>
#include <hip/hip_bf16.h>

// TriangleAttentionStartingNode: B=1, N=256, C=128, H=4, D=32
// bf16 WMMA (v_wmma_f32_16x16x32_bf16) for all GEMMs, fp32 softmax.
// Softmax runs in log2-domain: LOG2E folded into q-scale / mask / pair bias.
// Attention Q/K staged into LDS with gfx1250 async-to-LDS loads (ASYNCcnt).

#define NRES 256
#define CDIM 128
#define HEADS 4
#define DDIM 32
#define NROWS (NRES * NRES)          // 65536
#define QSCALE 0.17677669529663687f  // 1/sqrt(32)
#define LOG2E 1.4426950408889634f

typedef __bf16 v16bf __attribute__((ext_vector_type(16)));
typedef __bf16 v8bf  __attribute__((ext_vector_type(8)));
typedef float  v8f   __attribute__((ext_vector_type(8)));

// ---- bf16 helpers (storage = unsigned short, native HW converts) ---------
__device__ __forceinline__ unsigned short f2bfbits(float f) {
  union { __bf16 h; unsigned short s; } u; u.h = (__bf16)f;   // native fptrunc
  return u.s;
}
__device__ __forceinline__ float bfbits2f(unsigned short s) {
  union { unsigned u; float f; } out; out.u = ((unsigned)s) << 16;
  return out.f;
}

// ---- vectorized fragment loaders (contiguous k order, wave32) ------------
__device__ __forceinline__ v8bf ld8(const unsigned short* p) {
  return *reinterpret_cast<const v8bf*>(p);   // 16B-aligned by construction
}
__device__ __forceinline__ v16bf cat8(v8bf lo, v8bf hi) {
  return __builtin_shufflevector(lo, hi, 0, 1, 2, 3, 4, 5, 6, 7,
                                 8, 9, 10, 11, 12, 13, 14, 15);
}
// A (16x32 bf16, row-major src, leading dim ld, ld multiple of 8):
//   lanes 0-15: M=lane,   K in {0..15}
//   lanes16-31: M=lane-16,K in {16..31}
__device__ __forceinline__ v16bf load_a_bf(const unsigned short* __restrict__ src, int ld) {
  const int lane = threadIdx.x & 31;
  const unsigned short* p = src + (lane & 15) * ld + ((lane >> 4) << 4);
  return cat8(ld8(p), ld8(p + 8));
}
// B frag where source is B^T row-major (src[N][K]): frag[e] = src[n][kb+e],
// contiguous 16 elements per lane. n = lane&15, kb = 16*(lane>>4).
__device__ __forceinline__ v16bf load_b_contig(const unsigned short* __restrict__ src, int ld) {
  const int lane = threadIdx.x & 31;
  const unsigned short* p = src + (lane & 15) * ld + ((lane >> 4) << 4);
  return cat8(ld8(p), ld8(p + 8));
}
// Pre-packed fragment: 32 lanes x 16 elems contiguous
__device__ __forceinline__ v16bf load_frag_packed(const unsigned short* __restrict__ src) {
  const unsigned short* p = src + (threadIdx.x & 31) * 16;
  return cat8(ld8(p), ld8(p + 8));
}
__device__ __forceinline__ v8f wmma_bf(v16bf a, v16bf b, v8f c) {
  return __builtin_amdgcn_wmma_f32_16x16x32_bf16(false, a, false, b, (short)0, c, false, false);
}
// gfx1250 async global->LDS copy, 16B per lane, tracked by ASYNCcnt
__device__ __forceinline__ void async_copy16(unsigned lds_off, const void* gaddr) {
  asm volatile("global_load_async_to_lds_b128 %0, %1, off"
               :: "v"(lds_off), "v"(gaddr) : "memory");
}

// ---- Kernel 0: pack f32 weights into bf16 B-fragment order ---------------
__global__ void pack_w_kernel(const float* __restrict__ wq, const float* __restrict__ wk,
                              const float* __restrict__ wv, const float* __restrict__ wg,
                              const float* __restrict__ wo, unsigned short* __restrict__ pw) {
  const float* Ws[5] = {wq, wk, wv, wg, wo};
  const float* W = Ws[blockIdx.y];
  const int ts = blockIdx.x;                 // 0..31 : t = ts>>2, s = ts&3
  const int t = ts >> 2, s = ts & 3;
  const int lane = threadIdx.x & 31;
  const int n = lane & 15, kb = (lane >> 4) << 4;
  unsigned short* dst = pw + ((long)blockIdx.y * 32 + ts) * 512 + lane * 16;
#pragma unroll
  for (int e = 0; e < 16; ++e)
    dst[e] = f2bfbits(W[(s * 32 + kb + e) * CDIM + t * 16 + n]);
}

// ---- Kernel 1: LayerNorm over C, write bf16 ------------------------------
__global__ void ln_kernel(const float* __restrict__ act, const float* __restrict__ g,
                          const float* __restrict__ b, unsigned short* __restrict__ xb) {
  const int wave = threadIdx.x >> 5, lane = threadIdx.x & 31;
  const int row = blockIdx.x * 8 + wave;           // 0..65535
  const float* src = act + (long)row * CDIM;
  float v[4]; float s = 0.f;
#pragma unroll
  for (int e = 0; e < 4; ++e) { v[e] = src[lane + 32 * e]; s += v[e]; }
#pragma unroll
  for (int m = 16; m >= 1; m >>= 1) s += __shfl_xor(s, m, 32);
  const float mu = s * (1.f / 128.f);
  float var = 0.f;
#pragma unroll
  for (int e = 0; e < 4; ++e) { float d = v[e] - mu; var += d * d; }
#pragma unroll
  for (int m = 16; m >= 1; m >>= 1) var += __shfl_xor(var, m, 32);
  const float rs = rsqrtf(var * (1.f / 128.f) + 1e-5f);
#pragma unroll
  for (int e = 0; e < 4; ++e) {
    int c = lane + 32 * e;
    xb[(long)row * CDIM + c] = f2bfbits((v[e] - mu) * rs * g[c] + b[c]);
  }
}

// ---- Kernel 2: fused QKVG projection GEMM --------------------------------
// Wave caches 4 A-fragments and sweeps 4 column tiles (16 WMMAs, A loaded 1x).
__global__ __launch_bounds__(128) void proj_kernel(
    const unsigned short* __restrict__ xb, const unsigned short* __restrict__ pw,
    const float* __restrict__ bg, unsigned short* __restrict__ qkvg) {
  const int wave = threadIdx.x >> 5, lane = threadIdx.x & 31;
  const int wg = blockIdx.y * 4 + wave;   // 0..7: group of 4 col tiles
  const int which = wg >> 1;              // 0=q 1=k 2=v 3=g (wave-uniform)
  const int m0 = blockIdx.x * 16;
  const unsigned short* pwW = pw + (long)which * 32 * 512;
  unsigned short* dst = qkvg + (long)which * ((long)NROWS * CDIM);

  v16bf a4[4];
#pragma unroll
  for (int s = 0; s < 4; ++s)
    a4[s] = load_a_bf(xb + (long)m0 * CDIM + s * 32, CDIM);

  const float scale = (which == 0) ? (QSCALE * LOG2E) : 1.f;
  const bool isg = (which == 3);
  const int mrow = (lane >> 4) << 3;
#pragma unroll
  for (int tt = 0; tt < 4; ++tt) {
    const int t = (wg & 1) * 4 + tt;      // col tile 0..7 within this weight
    v8f acc = {0.f, 0.f, 0.f, 0.f, 0.f, 0.f, 0.f, 0.f};
#pragma unroll
    for (int s = 0; s < 4; ++s)
      acc = wmma_bf(a4[s], load_frag_packed(pwW + (t * 4 + s) * 512), acc);
    const int n = t * 16 + (lane & 15);
    const int h = n >> 5, d = n & 31;
    const float bgv = bg[h * DDIM + d];
#pragma unroll
    for (int r = 0; r < 8; ++r) {
      int m = m0 + r + mrow;
      int i = m >> 8, j = m & 255;
      float val = acc[r] * scale;
      if (isg)                            // uniform branch: gate sigmoid
        val = __builtin_amdgcn_rcpf(1.f + __builtin_amdgcn_exp2f(-(val + bgv) * LOG2E));
      dst[(((long)(i * HEADS + h)) * NRES + j) * DDIM + d] = f2bfbits(val);
    }
  }
}

// ---- Kernel 3: pair bias nb[h][q][k] = LOG2E * x[q,k,:].w2d[:,h] ---------
__global__ void nb_kernel(const unsigned short* __restrict__ xb,
                          const float* __restrict__ w2d, float* __restrict__ nb) {
  const int row = blockIdx.x * 128 + threadIdx.x;   // q*256+k
  float s0 = 0.f, s1 = 0.f, s2 = 0.f, s3 = 0.f;
  const uint4* xv = (const uint4*)(xb + (long)row * CDIM);
#pragma unroll 4
  for (int c8 = 0; c8 < 16; ++c8) {
    union { uint4 q; unsigned short s[8]; } u; u.q = xv[c8];
#pragma unroll
    for (int j = 0; j < 8; ++j) {
      const float xval = bfbits2f(u.s[j]);
      const int c = c8 * 8 + j;
      s0 += xval * w2d[c * 4 + 0]; s1 += xval * w2d[c * 4 + 1];
      s2 += xval * w2d[c * 4 + 2]; s3 += xval * w2d[c * 4 + 3];
    }
  }
  const int q = row >> 8, k = row & 255;
  nb[(0 * NRES + q) * NRES + k] = s0 * LOG2E;
  nb[(1 * NRES + q) * NRES + k] = s1 * LOG2E;
  nb[(2 * NRES + q) * NRES + k] = s2 * LOG2E;
  nb[(3 * NRES + q) * NRES + k] = s3 * LOG2E;
}

// ---- Kernel 4: attention per (i,h). 8 waves; wave owns 32 q-rows (2 x 16) ---------
__global__ __launch_bounds__(256, 1) void attn_kernel(
    const unsigned short* __restrict__ qg, const unsigned short* __restrict__ kg,
    const unsigned short* __restrict__ vg, const unsigned short* __restrict__ gg,
    const float* __restrict__ nb, const float* __restrict__ mask,
    unsigned short* __restrict__ ob) {
  extern __shared__ char smem[];
  unsigned short* Qs = (unsigned short*)smem;              // 256x32 bf16
  unsigned short* Ks = Qs + NRES * DDIM;                   // 256x32 bf16
  unsigned short* Vt = Ks + NRES * DDIM;                   // 32x256 bf16 (transposed V)
  float* maskb = (float*)(Vt + DDIM * NRES);               // 256 f32
  unsigned short* probsAll = (unsigned short*)(maskb + NRES); // 8 * 16*256 bf16

  const int i = blockIdx.x, h = blockIdx.y;
  const int tid = threadIdx.x, wave = tid >> 5, lane = tid & 31;
  const long base = ((long)(i * HEADS + h)) * NRES * DDIM;

  { // Q,K: async global->LDS (no VGPR transit, ASYNCcnt-tracked)
    const unsigned short* gq = qg + base;
    const unsigned short* gk = kg + base;
    const unsigned qoff = (unsigned)(size_t)Qs;
    const unsigned koff = (unsigned)(size_t)Ks;
    for (int t = tid; t < NRES * DDIM / 8; t += 256) {
      async_copy16(qoff + t * 16, gq + t * 8);
      async_copy16(koff + t * 16, gk + t * 8);
    }
    // V transposed (Vt[d][k]) via VGPR (transpose can't be a straight copy)
    const uint4* vsrc = (const uint4*)(vg + base + (long)tid * DDIM);  // thread = one k row
#pragma unroll
    for (int c = 0; c < 4; ++c) {
      union { uint4 q; unsigned short s[8]; } u; u.q = vsrc[c];
#pragma unroll
      for (int j = 0; j < 8; ++j) Vt[(c * 8 + j) * NRES + tid] = u.s[j];
    }
    if (tid < NRES) maskb[tid] = (1e9f * LOG2E) * (mask[i * NRES + tid] - 1.f);
  }
  asm volatile("s_wait_asynccnt 0" ::: "memory");
  __syncthreads();

  unsigned short* probs = probsAll + wave * 16 * NRES;
  const int lg = lane >> 4, nn = lane & 15;

  for (int rbi = 0; rbi < 2; ++rbi) {
    const int q0 = (wave * 2 + rbi) * 16;

    // ---- logits: Q[16x32] x K^T  ->  16 tiles of 16x16 f32 -------------
    v8f acc[16];
#pragma unroll
    for (int t = 0; t < 16; ++t)
#pragma unroll
      for (int r = 0; r < 8; ++r) acc[t][r] = 0.f;

    const v16bf aq = load_a_bf(Qs + q0 * DDIM, DDIM);
#pragma unroll
    for (int kt = 0; kt < 16; ++kt) {
      v16bf bk = load_b_contig(Ks + kt * 16 * DDIM, DDIM);   // B[d][k] = K[k][d]
      acc[kt] = wmma_bf(aq, bk, acc[kt]);
    }

    // ---- add mask bias + pair bias (both pre-scaled by LOG2E) ----------
#pragma unroll
    for (int kt = 0; kt < 16; ++kt) {
      const int k = kt * 16 + nn;
      const float mb = maskb[k];
#pragma unroll
      for (int r = 0; r < 8; ++r) {
        const int m = q0 + r + (lg << 3);
        acc[kt][r] += mb + nb[((long)(h * NRES + m)) * NRES + k];
      }
    }

    // ---- in-register softmax over k (log2 domain: exp2, hw rcp) --------
#pragma unroll
    for (int r = 0; r < 8; ++r) {
      float mx = -1e30f;
#pragma unroll
      for (int kt = 0; kt < 16; ++kt) mx = fmaxf(mx, acc[kt][r]);
#pragma unroll
      for (int s = 8; s >= 1; s >>= 1) mx = fmaxf(mx, __shfl_xor(mx, s, 32));
      float sum = 0.f;
#pragma unroll
      for (int kt = 0; kt < 16; ++kt) {
        float e = __builtin_amdgcn_exp2f(acc[kt][r] - mx);
        acc[kt][r] = e; sum += e;
      }
#pragma unroll
      for (int s = 8; s >= 1; s >>= 1) sum += __shfl_xor(sum, s, 32);
      const float inv = __builtin_amdgcn_rcpf(sum);
      const int m = r + (lg << 3);     // row within block (0..15)
#pragma unroll
      for (int kt = 0; kt < 16; ++kt)
        probs[m * NRES + kt * 16 + nn] = f2bfbits(acc[kt][r] * inv);
    }

    // ---- o = A[16x256] x V[256x32], then gate + store ------------------
#pragma unroll
    for (int dt = 0; dt < 2; ++dt) {
      v8f o = {0.f, 0.f, 0.f, 0.f, 0.f, 0.f, 0.f, 0.f};
#pragma unroll
      for (int ks = 0; ks < 8; ++ks) {
        v16bf ap = load_a_bf(probs + ks * 32, NRES);
        v16bf bv = load_b_contig(Vt + dt * 16 * NRES + ks * 32, NRES);  // B[k][d]=Vt[d][k]^T
        o = wmma_bf(ap, bv, o);
      }
      const int d = dt * 16 + nn;
#pragma unroll
      for (int r = 0; r < 8; ++r) {
        const int m = q0 + r + (lg << 3);
        const float gv = bfbits2f(gg[base + (long)m * DDIM + d]);
        ob[((long)i * NRES + m) * CDIM + h * DDIM + d] = f2bfbits(o[r] * gv);
      }
    }
  }
}

// ---- Kernel 5: out = (o*g)[65536x128] x wo[128x128] + bo -----------------
// 2 waves/block, each sweeps 4 col tiles with cached A fragments.
__global__ __launch_bounds__(64) void out_kernel(
    const unsigned short* __restrict__ ob, const unsigned short* __restrict__ pwo,
    const float* __restrict__ bo, float* __restrict__ out) {
  const int wave = threadIdx.x >> 5, lane = threadIdx.x & 31;
  const int m0 = blockIdx.x * 16;
  v16bf a4[4];
#pragma unroll
  for (int s = 0; s < 4; ++s)
    a4[s] = load_a_bf(ob + (long)m0 * CDIM + s * 32, CDIM);
  const int mrow = (lane >> 4) << 3;
#pragma unroll
  for (int tt = 0; tt < 4; ++tt) {
    const int ct = wave * 4 + tt;         // 0..7
    v8f acc = {0.f, 0.f, 0.f, 0.f, 0.f, 0.f, 0.f, 0.f};
#pragma unroll
    for (int s = 0; s < 4; ++s)
      acc = wmma_bf(a4[s], load_frag_packed(pwo + (ct * 4 + s) * 512), acc);
    const int n = ct * 16 + (lane & 15);
    const float bias = bo[n];
#pragma unroll
    for (int r = 0; r < 8; ++r) {
      const int m = m0 + r + mrow;
      out[(long)m * CDIM + n] = acc[r] + bias;
    }
  }
}

extern "C" void kernel_launch(void* const* d_in, const int* in_sizes, int n_in,
                              void* d_out, int out_size, void* d_ws, size_t ws_size,
                              hipStream_t stream) {
  (void)in_sizes; (void)n_in; (void)out_size; (void)ws_size;
  const float* act  = (const float*)d_in[0];
  const float* mask = (const float*)d_in[1];
  const float* ln_g = (const float*)d_in[2];
  const float* ln_b = (const float*)d_in[3];
  const float* w2d  = (const float*)d_in[4];
  const float* wq   = (const float*)d_in[5];
  const float* wk   = (const float*)d_in[6];
  const float* wv   = (const float*)d_in[7];
  const float* wg   = (const float*)d_in[8];
  const float* bg   = (const float*)d_in[9];
  const float* wo   = (const float*)d_in[10];
  const float* bo   = (const float*)d_in[11];
  float* out = (float*)d_out;

  char* ws = (char*)d_ws;
  const size_t MB = 1024 * 1024;
  unsigned short* xb = (unsigned short*)(ws);              // 16MB (reused as ob)
  unsigned short* qb = (unsigned short*)(ws + 16 * MB);    // qkvg base (4 x 16MB)
  unsigned short* kb = (unsigned short*)(ws + 32 * MB);
  unsigned short* vb = (unsigned short*)(ws + 48 * MB);
  unsigned short* gb = (unsigned short*)(ws + 64 * MB);
  float*          nb = (float*)(ws + 80 * MB);             // 1MB
  unsigned short* pw = (unsigned short*)(ws + 81 * MB);    // 5 x 32KB packed weights
  unsigned short* obuf = xb;                               // reuse x region

  pack_w_kernel<<<dim3(32, 5), 32, 0, stream>>>(wq, wk, wv, wg, wo, pw);
  ln_kernel<<<NROWS / 8, 256, 0, stream>>>(act, ln_g, ln_b, xb);
  proj_kernel<<<dim3(NROWS / 16, 2), 128, 0, stream>>>(xb, pw, bg, qb);
  nb_kernel<<<NROWS / 128, 128, 0, stream>>>(xb, w2d, nb);

  const int attn_lds = NRES * DDIM * 2 * 3 + NRES * 4 + 8 * 16 * NRES * 2; // 115712 B (<320KB/WGP)
  attn_kernel<<<dim3(NRES, HEADS), 256, attn_lds, stream>>>(qb, kb, vb, gb, nb, mask, obuf);

  out_kernel<<<NROWS / 16, 64, 0, stream>>>(obuf, pw + 4 * 32 * 512, bo, out);
}
